// Hamming1Layer_83116207112786
// MI455X (gfx1250) — matrix operand order repaired and checked by reading, not compiled, not run on hardware
//
#include <hip/hip_runtime.h>

typedef __attribute__((ext_vector_type(16))) _Float16 v16h;
typedef __attribute__((ext_vector_type(8)))  float    v8f;
typedef __attribute__((ext_vector_type(4)))  float    f32x4;
typedef __attribute__((ext_vector_type(4)))  unsigned int u32x4;
typedef __attribute__((ext_vector_type(4)))  int      i32x4;
typedef __attribute__((ext_vector_type(8)))  int      i32x8;

#ifndef __has_builtin
#define __has_builtin(x) 0
#endif

#if __has_builtin(__builtin_amdgcn_tensor_load_to_lds) && __has_builtin(__builtin_amdgcn_s_wait_tensorcnt)
#define USE_TDM 1
#else
#define USE_TDM 0
#endif

namespace {
constexpr int kNBits = 14;
constexpr int kL     = 1 << kNBits;   // 16384
constexpr int kB     = 32;
constexpr int kCin   = 32;
constexpr int kCout  = 64;
constexpr int kTileN = 64;            // aligned l-tile: bits 0..5 are tile-local
constexpr int kPadC  = kCin + 8;
}

__global__ __launch_bounds__(128)
void hamming_mix_wmma(const float* __restrict__ x,
                      const float* __restrict__ w_self_p,
                      const float* __restrict__ w_bits,
                      const float* __restrict__ mix_w,
                      const float* __restrict__ mix_b,
                      float* __restrict__ out)
{
    __shared__ __align__(16) float    x_lds[kCin][kTileN];  // self tile, 8 KB
    __shared__ __align__(16) _Float16 y_lds[kTileN][kPadC]; // blended tile (f16)

    const int tiles_per_b = kL / kTileN;        // 256
    const int b  = blockIdx.x / tiles_per_b;
    const int t  = blockIdx.x % tiles_per_b;
    const int l0 = t * kTileN;

    const float* xb = x + (size_t)b * kCin * kL;

    // ---- Stage self tile (32 x 64 f32, row stride L) into LDS ----
#if USE_TDM
    if (threadIdx.x == 0) {
        // Tensor DMA descriptor (D#), normal mode, 2D tile.
        const unsigned long long ga =
            (unsigned long long)(uintptr_t)(xb + l0) & 0x1FFFFFFFFFFFFFull;
        const unsigned int lds_addr = (unsigned int)(uintptr_t)&x_lds[0][0];

        u32x4 g0;
        g0[0] = 1u;                                   // count=1, valid user D#
        g0[1] = lds_addr;                             // lds_addr [63:32]
        g0[2] = (unsigned int)ga;                     // global_addr[31:0]
        g0[3] = (unsigned int)((ga >> 32) & 0x1FFFFFFu) | (2u << 30); // +type=2

        i32x8 g1;
        g1[0] = (int)(2u << 16);                      // data_size=4B, mask=0
        g1[1] = (int)((unsigned)kL << 16);            // tensor_dim0[15:0] @ [63:48]
        g1[2] = (int)((unsigned)kCin << 16);          // dim0 hi=0 | tensor_dim1 lo
        g1[3] = (int)((unsigned)kTileN << 16);        // tensor_dim1 hi=0 | tile_dim0
        g1[4] = (int)(unsigned)kCin;                  // tile_dim1=32, tile_dim2=0
        g1[5] = (int)(unsigned)kL;                    // tensor_dim0_stride[31:0]
        g1[6] = 0;                                    // stride hi / dim1_stride lo
        g1[7] = 0;

        i32x4 gz = {0, 0, 0, 0};
#if defined(__clang_major__) && (__clang_major__ >= 23)
        i32x8 gz8 = {0, 0, 0, 0, 0, 0, 0, 0};
        __builtin_amdgcn_tensor_load_to_lds(g0, g1, gz, gz, gz8, 0);
#else
        __builtin_amdgcn_tensor_load_to_lds(g0, g1, gz, gz, 0);
#endif
        __builtin_amdgcn_s_wait_tensorcnt(0);
    }
#else
    for (int idx = threadIdx.x; idx < kCin * (kTileN / 4); idx += blockDim.x) {
        const int c  = idx >> 4;
        const int gi = idx & 15;
        *(f32x4*)&x_lds[c][gi * 4] =
            *(const f32x4*)(xb + (size_t)c * kL + l0 + gi * 4);
    }
#endif
    __syncthreads();

    const float w_self = w_self_p[0];
    float wb[kNBits];
    #pragma unroll
    for (int k = 0; k < kNBits; ++k) wb[k] = w_bits[k];
    const float inv = 1.0f / (1.0f + (float)kNBits);

    // ---- Phase 1: hypercube blend. bits 0-1 in-register, 2-5 from LDS,
    //      6-13 as vector b128 gathers from global (L2-resident). ----
    for (int idx = threadIdx.x; idx < kCin * (kTileN / 4); idx += blockDim.x) {
        const int c   = idx >> 4;       // channel
        const int gi  = idx & 15;       // float4 group within tile
        const int li4 = gi * 4;         // local l of group base

        const f32x4 s = *(const f32x4*)&x_lds[c][li4];

        f32x4 lk[4];
        #pragma unroll
        for (int k = 2; k < 6; ++k) {
            lk[k - 2] = *(const f32x4*)&x_lds[c][li4 ^ (1 << k)];
        }

        const float* xc = xb + (size_t)c * kL;
        f32x4 rk[8];
        #pragma unroll
        for (int k = 6; k < kNBits; ++k) {
            rk[k - 6] = *(const f32x4*)(xc + ((l0 + li4) ^ (1 << k)));
        }

        float acc[4];
        #pragma unroll
        for (int j = 0; j < 4; ++j) {
            acc[j] = w_self * s[j] + wb[0] * s[j ^ 1] + wb[1] * s[j ^ 2];
        }
        #pragma unroll
        for (int k = 2; k < 6; ++k) {
            #pragma unroll
            for (int j = 0; j < 4; ++j) acc[j] += wb[k] * lk[k - 2][j];
        }
        #pragma unroll
        for (int k = 6; k < kNBits; ++k) {
            #pragma unroll
            for (int j = 0; j < 4; ++j) acc[j] += wb[k] * rk[k - 6][j];
        }
        #pragma unroll
        for (int j = 0; j < 4; ++j) {
            y_lds[li4 + j][c] = (_Float16)(acc[j] * inv);
        }
    }
    __syncthreads();

    // ---- Phase 2: out(64 x l) = mix_w(64x32) @ y(32 x l) via WMMA f16 ----
    const int lane = threadIdx.x & 31;
    const int wave = threadIdx.x >> 5;          // M-tile
    const int g    = lane >> 4;
    const int hr   = lane & 15;

    const int mrow = wave * 16 + hr;
    v16h a;
    #pragma unroll
    for (int e = 0; e < 16; ++e) {
        const int K = (e < 8 ? e : e + 8) + 8 * g;   // ISA 16-bit A 16x32 map
        a[e] = (_Float16)mix_w[mrow * kCin + K];
    }

    float bias[8];
    #pragma unroll
    for (int r = 0; r < 8; ++r) bias[r] = mix_b[wave * 16 + r + 8 * g];

    float* outb = out + (size_t)b * kCout * kL;

    #pragma unroll
    for (int nt = 0; nt < kTileN / 16; ++nt) {
        const int ln = nt * 16 + hr;

        v16h bf;
        #pragma unroll
        for (int e = 0; e < 16; ++e) bf[e] = y_lds[ln][(e < 8 ? e : e + 8) + 8 * g];

        v8f c = {};
        c = __builtin_amdgcn_wmma_f32_16x16x32_f16(
                false, a, false, bf, (short)0, c, false, false);

        #pragma unroll
        for (int r = 0; r < 8; ++r) {
            const int o = wave * 16 + r + 8 * g;
            outb[(size_t)o * kL + l0 + ln] = c[r] + bias[r];
        }
    }
}

extern "C" void kernel_launch(void* const* d_in, const int* in_sizes, int n_in,
                              void* d_out, int out_size, void* d_ws, size_t ws_size,
                              hipStream_t stream) {
    (void)in_sizes; (void)n_in; (void)out_size; (void)d_ws; (void)ws_size;
    const float* x      = (const float*)d_in[0];
    const float* w_self = (const float*)d_in[1];
    const float* w_bits = (const float*)d_in[2];
    const float* mix_w  = (const float*)d_in[3];
    const float* mix_b  = (const float*)d_in[4];
    float* out = (float*)d_out;

    dim3 grid(kB * (kL / kTileN));   // 8192 blocks
    dim3 block(128);                 // 4 waves (wave32)
    hipLaunchKernelGGL(hamming_mix_wmma, grid, block, 0, stream,
                       x, w_self, w_bits, mix_w, mix_b, out);
}
